// GraphConvolutionSparse_53154515256136
// MI455X (gfx1250) — matrix operand order repaired and checked by reading, not compiled, not run on hardware
//
#include <hip/hip_runtime.h>
#include <hip/hip_bf16.h>

#define N_NODES 100000
#define IN_DIM  256
#define OUT_DIM 128

typedef __attribute__((ext_vector_type(16))) __bf16 v16bf;
typedef __attribute__((ext_vector_type(8)))  float  v8f;

// ---------------- zero fill ----------------
__global__ void zero_f32_kernel(float* __restrict__ p, int n) {
    int i = blockIdx.x * blockDim.x + threadIdx.x;
    if (i < n) p[i] = 0.0f;
}

// ---------------- densify sparse X (COO -> dense f32) ----------------
__global__ void scatter_feat_kernel(const float* __restrict__ vals,
                                    const int* __restrict__ rows,
                                    const int* __restrict__ cols,
                                    float* __restrict__ Xd, int nnz) {
    int i = blockIdx.x * blockDim.x + threadIdx.x;
    if (i < nnz) {
        atomicAdd(&Xd[(size_t)rows[i] * IN_DIM + cols[i]], vals[i]);
    }
}

// ---------------- pre-swizzle W into chunk-major WMMA B-fragment layout (bf16 hi/lo) ----
// Byte layout: Wfrag[t][plane][u][lane][e] ->
//   byte_off = t*16384 + plane*8192 + (u*32 + lane)*32  (+ 2*e)
// B fragment mapping this code uses (contiguous K order, matching A): lane l -> N = 16u + (l&15),
// element e -> K = 32t + 16*(l>>4) + e.
__global__ void convert_w_kernel(const float* __restrict__ W,
                                 __bf16* __restrict__ Wfrag) {
    int tid = blockIdx.x * blockDim.x + threadIdx.x;   // (t,u,lane): t*256 + u*32 + lane
    if (tid >= 8 * 8 * 32) return;
    int t    = tid >> 8;
    int u    = (tid >> 5) & 7;
    int lane = tid & 31;
    int n     = 16 * u + (lane & 15);
    int kbase = 32 * t + 16 * (lane >> 4);
    __bf16* ph = Wfrag + (size_t)t * 8192 + (size_t)(u * 32 + lane) * 16;          // hi plane
    __bf16* pl = ph + 4096;                                                        // lo plane
#pragma unroll
    for (int e = 0; e < 16; ++e) {
        float w = W[(size_t)(kbase + e) * OUT_DIM + n];
        __bf16 h = (__bf16)w;
        ph[e] = h;
        pl[e] = (__bf16)(w - (float)h);
    }
}

// ---------------- XW = Xdense @ W via WMMA, B staged in LDS with async copies ----------
// Block = 256 threads = 8 waves; wave w computes rows [blockIdx.x*128 + 16w, +16) x 128 cols.
// Double-buffered 2 x 16KB LDS; each wave issues 4 global_load_async_to_lds_b128 per chunk.
__global__ __launch_bounds__(256)
void gemm_xw_kernel(const float* __restrict__ Xd,
                    const __bf16* __restrict__ Wfrag,
                    float* __restrict__ XW) {
    __shared__ v16bf smemv[1024];            // 32 KB = 2 phases x 16 KB
    char* smem = (char*)smemv;
    const char* wbytes = (const char*)Wfrag;

    const int lane    = threadIdx.x & 31;
    const int wave    = threadIdx.x >> 5;
    const int rowBase = blockIdx.x * 128 + wave * 16;
    const int m       = lane & 15;
    const int khalf   = lane >> 4;          // 0 or 1
    const int row     = rowBase + m;        // A-layout row for this lane
    const float scale = (row < N_NODES) ? 1.0f : 0.0f;   // value-mask OOB rows (EXEC stays all-1s)
    const int srcRow  = (row < N_NODES) ? row : 0;
    const float* __restrict__ xrow = Xd + (size_t)srcRow * IN_DIM;

    // async-stage one 16KB W chunk into LDS phase buffer (4 x 512B per wave, 16B per lane)
    auto stage = [&](int t, int phase) {
#pragma unroll
        for (int j = 0; j < 4; ++j) {
            int off = wave * 2048 + j * 512 + lane * 16;
            unsigned laddr = (unsigned)(size_t)(smem + phase * 16384 + off);
            unsigned long long gaddr =
                (unsigned long long)(size_t)(wbytes + (size_t)t * 16384 + off);
            asm volatile("global_load_async_to_lds_b128 %0, %1, off"
                         :: "v"(laddr), "v"(gaddr) : "memory");
        }
    };

    stage(0, 0);                             // prologue

    v8f acc[8] = {};                         // 8 N-tiles of 16x16 f32

    for (int t = 0; t < 8; ++t) {            // K chunks of 32
        asm volatile("s_wait_asynccnt 0x0" ::: "memory");   // my copies for chunk t done
        __syncthreads();                     // everyone's copies done; other phase free
        if (t < 7) stage(t + 1, (t + 1) & 1);
        if (t < 7) __builtin_prefetch(xrow + 32 * (t + 1) + 16 * khalf, 0, 3);

        // A fragment (contiguous K order, matching Wfrag): e -> K=16*khalf+e
        const float4 q0 = *(const float4*)(xrow + 32 * t + 16 * khalf);
        const float4 q1 = *(const float4*)(xrow + 32 * t + 16 * khalf + 4);
        const float4 q2 = *(const float4*)(xrow + 32 * t + 16 * khalf + 8);
        const float4 q3 = *(const float4*)(xrow + 32 * t + 16 * khalf + 12);
        float af[16] = { q0.x, q0.y, q0.z, q0.w,  q1.x, q1.y, q1.z, q1.w,
                         q2.x, q2.y, q2.z, q2.w,  q3.x, q3.y, q3.z, q3.w };
        v16bf ahi, alo;
#pragma unroll
        for (int e = 0; e < 16; ++e) {
            float x  = af[e] * scale;
            __bf16 h = (__bf16)x;
            ahi[e] = h;
            alo[e] = (__bf16)(x - (float)h);
        }

        const char* sb = smem + (t & 1) * 16384;
#pragma unroll
        for (int u = 0; u < 8; ++u) {        // 8 N tiles of 16 cols
            const v16bf bh = *(const v16bf*)(sb + (size_t)(u * 32 + lane) * 32);
            const v16bf bl = *(const v16bf*)(sb + 8192 + (size_t)(u * 32 + lane) * 32);
            acc[u] = __builtin_amdgcn_wmma_f32_16x16x32_bf16(false, ahi, false, bh, (short)0, acc[u], false, false);
            acc[u] = __builtin_amdgcn_wmma_f32_16x16x32_bf16(false, alo, false, bh, (short)0, acc[u], false, false);
            acc[u] = __builtin_amdgcn_wmma_f32_16x16x32_bf16(false, ahi, false, bl, (short)0, acc[u], false, false);
        }
    }

    // C/D layout: VGPR r at lane l -> row = r + 8*(l>>4), col = l&15
#pragma unroll
    for (int r = 0; r < 8; ++r) {
        const int rowOut = rowBase + r + 8 * khalf;
        if (rowOut < N_NODES) {
            float* __restrict__ orow = XW + (size_t)rowOut * OUT_DIM + (lane & 15);
#pragma unroll
            for (int u = 0; u < 8; ++u) orow[u * 16] = acc[u][r];
        }
    }
}

// ---------------- out += adj_val * XW[col]  (one wave per nnz, L2-resident atomics) -----
__global__ __launch_bounds__(256)
void spmm_adj_kernel(const float* __restrict__ vals,
                     const int* __restrict__ rows,
                     const int* __restrict__ cols,
                     const float* __restrict__ XW,
                     float* __restrict__ out, int nnz) {
    int wid  = (blockIdx.x * blockDim.x + threadIdx.x) >> 5;
    int lane = threadIdx.x & 31;
    if (wid >= nnz) return;
    float v = vals[wid];
    int r = rows[wid];
    int c = cols[wid];
    const float4 x = *(const float4*)(XW + (size_t)c * OUT_DIM + lane * 4);
    float* o = out + (size_t)r * OUT_DIM + lane * 4;
    atomicAdd(o + 0, v * x.x);
    atomicAdd(o + 1, v * x.y);
    atomicAdd(o + 2, v * x.z);
    atomicAdd(o + 3, v * x.w);
}

// ---------------- bias + ReLU ----------------
__global__ void bias_relu_kernel(float* __restrict__ out, const float* __restrict__ bias, int n) {
    int i = blockIdx.x * blockDim.x + threadIdx.x;
    if (i < n) {
        float v = out[i] + bias[i & (OUT_DIM - 1)];
        out[i] = v > 0.0f ? v : 0.0f;
    }
}

extern "C" void kernel_launch(void* const* d_in, const int* in_sizes, int n_in,
                              void* d_out, int out_size, void* d_ws, size_t ws_size,
                              hipStream_t stream) {
    const float* feat_vals = (const float*)d_in[0];
    const int*   feat_rows = (const int*)d_in[1];
    const int*   feat_cols = (const int*)d_in[2];
    const float* adj_vals  = (const float*)d_in[3];
    const int*   adj_rows  = (const int*)d_in[4];
    const int*   adj_cols  = (const int*)d_in[5];
    const float* weights   = (const float*)d_in[6];
    const float* bias      = (const float*)d_in[7];
    float*       out       = (float*)d_out;

    const int fnnz = in_sizes[0];
    const int annz = in_sizes[3];

    // Workspace layout:
    //   Xd    : dense X,  f32 [N_NODES][IN_DIM]      = 102,400,000 B
    //   XW    : X@W,      f32 [N_NODES][OUT_DIM]     =  51,200,000 B
    //   Wfrag : bf16 hi/lo fragments, chunk-major    =     131,072 B
    char*   ws    = (char*)d_ws;
    float*  Xd    = (float*)ws;
    float*  XW    = (float*)(ws + (size_t)N_NODES * IN_DIM * sizeof(float));
    __bf16* Wfrag = (__bf16*)(ws + (size_t)N_NODES * IN_DIM * sizeof(float)
                                 + (size_t)N_NODES * OUT_DIM * sizeof(float));

    // 1) zero dense X and output accumulator
    {
        int n = N_NODES * IN_DIM;
        zero_f32_kernel<<<(n + 255) / 256, 256, 0, stream>>>(Xd, n);
    }
    {
        int n = N_NODES * OUT_DIM;
        zero_f32_kernel<<<(n + 255) / 256, 256, 0, stream>>>(out, n);
    }

    // 2) densify sparse features
    scatter_feat_kernel<<<(fnnz + 255) / 256, 256, 0, stream>>>(
        feat_vals, feat_rows, feat_cols, Xd, fnnz);

    // 3) pre-swizzle W into chunk-major WMMA B fragments (hi/lo bf16 split)
    convert_w_kernel<<<(8 * 8 * 32 + 255) / 256, 256, 0, stream>>>(weights, Wfrag);

    // 4) XW = Xd @ W via v_wmma_f32_16x16x32_bf16, W async-staged through LDS
    gemm_xw_kernel<<<(N_NODES + 127) / 128, 256, 0, stream>>>(Xd, Wfrag, XW);

    // 5) out += A * XW  (wave per nnz)
    {
        long threads = (long)annz * 32;
        int blocks = (int)((threads + 255) / 256);
        spmm_adj_kernel<<<blocks, 256, 0, stream>>>(adj_vals, adj_rows, adj_cols, XW, out, annz);
    }

    // 6) bias + ReLU
    {
        int n = N_NODES * OUT_DIM;
        bias_relu_kernel<<<(n + 255) / 256, 256, 0, stream>>>(out, bias, n);
    }
}
